// MultiHeadedAttention_v2_54631984005574
// MI455X (gfx1250) — matrix operand order repaired and checked by reading, not compiled, not run on hardware
//
#include <hip/hip_runtime.h>
#include <stdint.h>

// ---------------- CDNA5 WMMA plumbing ----------------
typedef __attribute__((ext_vector_type(16))) __bf16 v16bf;
typedef __attribute__((ext_vector_type(2)))  __bf16 v2bf;
typedef __attribute__((ext_vector_type(8)))  float  v8f;

union FragAB { v16bf v; uint32_t u[8]; uint4 q[2]; };

__device__ __forceinline__ v8f wmma_bf16(const FragAB& a, const FragAB& b, v8f c) {
  return __builtin_amdgcn_wmma_f32_16x16x32_bf16(false, a.v, false, b.v,
                                                 (short)0, c, false, false);
}

// f32 -> bf16 with round-half-up (ties differ from RNE only on exact ties).
__device__ __forceinline__ uint16_t f32_to_bf16(float f) {
  return (uint16_t)((__float_as_uint(f) + 0x8000u) >> 16);
}
// Packed pair: 2x v_add + 1x v_perm_b32.
__device__ __forceinline__ uint32_t pack_bf16(float lo, float hi) {
#if __has_builtin(__builtin_amdgcn_cvt_pk_bf16_f32)
  union { v2bf v; uint32_t u; } cv;
  cv.v = __builtin_amdgcn_cvt_pk_bf16_f32(lo, hi);
  return cv.u;
#else
  uint32_t ul = __float_as_uint(lo) + 0x8000u;
  uint32_t uh = __float_as_uint(hi) + 0x8000u;
  return __builtin_amdgcn_perm(uh, ul, 0x07060302u);  // {uh[31:16], ul[31:16]}
#endif
}

#define KC    32
#define KPAD  40   // row stride in ushorts: 80B, 16B-aligned, conflict-free

// Fragment gathers as 2 x b128 per the CDNA5 16-bit VGPR layouts.
__device__ __forceinline__ void load_fragA(FragAB& a, const uint16_t* row, int half) {
  const uint4* r = (const uint4*)row;
  a.q[0] = r[half];
  a.q[1] = r[2 + half];
}
__device__ __forceinline__ void load_fragB(FragAB& b, const uint16_t* rowT, int half) {
  const uint4* r = (const uint4*)rowT;
  b.q[0] = r[half];
  b.q[1] = r[2 + half];
}

// ---------------------------------------------------------------------------
// Kernel 1: Y(bf16)[4096,1024] = X(f32) @ W(f32) + bias
// 64x64 tile/WG, double-buffered LDS staging, last iteration peeled
// ---------------------------------------------------------------------------
#define PK 1024
#define PN 1024
#define PNCH (PK / KC)   // 32

__global__ __launch_bounds__(256) void proj_gemm(
    const float* __restrict__ X, const float* __restrict__ W,
    const float* __restrict__ bias, uint16_t* __restrict__ Y)
{
  __shared__ uint16_t ldsA[2][64][KPAD];
  __shared__ uint16_t ldsBt[2][64][KPAD];
  const int tid = threadIdx.x;
  const int w = tid >> 5, lane = tid & 31;
  const int half = lane >> 4, l15 = lane & 15;
  const int bm = blockIdx.x * 64, bn = blockIdx.y * 64;
  const int mt = w & 3, np = w >> 2;

  const int sm = tid >> 2, skq = (tid & 3) * 8;   // A-staging mapping

  auto stage = [&](int k0, int buf) {
    {
      const float4* src = (const float4*)&X[(size_t)(bm + sm) * PK + k0 + skq];
      float4 x0 = src[0], x1 = src[1];
      uint4 p = { pack_bf16(x0.x, x0.y), pack_bf16(x0.z, x0.w),
                  pack_bf16(x1.x, x1.y), pack_bf16(x1.z, x1.w) };
      *(uint4*)&ldsA[buf][sm][skq] = p;
    }
#pragma unroll
    for (int it = 0; it < 2; ++it) {
      int idx = tid + it * 256;
      int k = idx >> 4, nq = (idx & 15) * 4;
      float4 wv = *(const float4*)&W[(size_t)(k0 + k) * PN + bn + nq];
      uint32_t p01 = pack_bf16(wv.x, wv.y), p23 = pack_bf16(wv.z, wv.w);
      ldsBt[buf][nq + 0][k] = (uint16_t)p01;
      ldsBt[buf][nq + 1][k] = (uint16_t)(p01 >> 16);
      ldsBt[buf][nq + 2][k] = (uint16_t)p23;
      ldsBt[buf][nq + 3][k] = (uint16_t)(p23 >> 16);
    }
  };

  v8f acc0 = {}; v8f acc1 = {};
  stage(0, 0);
  __syncthreads();
  for (int c = 0; c < PNCH - 1; ++c) {       // steady state: unconditional
    const int buf = c & 1;
    FragAB a, b0, b1;                        // frag ds_loads first
    load_fragA(a, &ldsA[buf][mt * 16 + l15][0], half);
    load_fragB(b0, &ldsBt[buf][(np * 2 + 0) * 16 + l15][0], half);
    load_fragB(b1, &ldsBt[buf][(np * 2 + 1) * 16 + l15][0], half);
    stage((c + 1) * KC, buf ^ 1);            // overlap staging w/ WMMA
    acc0 = wmma_bf16(a, b0, acc0);
    acc1 = wmma_bf16(a, b1, acc1);
    __syncthreads();
  }
  {                                          // peeled last chunk
    const int buf = (PNCH - 1) & 1;
    FragAB a, b0, b1;
    load_fragA(a, &ldsA[buf][mt * 16 + l15][0], half);
    load_fragB(b0, &ldsBt[buf][(np * 2 + 0) * 16 + l15][0], half);
    load_fragB(b1, &ldsBt[buf][(np * 2 + 1) * 16 + l15][0], half);
    acc0 = wmma_bf16(a, b0, acc0);
    acc1 = wmma_bf16(a, b1, acc1);
  }
#pragma unroll
  for (int t = 0; t < 2; ++t) {
    v8f acc = (t == 0) ? acc0 : acc1;
    int n = bn + (np * 2 + t) * 16 + l15;
    float bv = bias[n];
#pragma unroll
    for (int r = 0; r < 8; ++r) {
      int m = bm + mt * 16 + r + 8 * half;
      Y[(size_t)m * PN + n] = f32_to_bf16(acc[r] + bv);
    }
  }
}

// ---------------------------------------------------------------------------
// Kernel 2: attention for one (h*4+b, 16-row t1 block)
// Dynamic LDS (~140KB/WG; 320KB/WGP -> 2 WGs/WGP)
// ---------------------------------------------------------------------------
#define SSROW 1032
#define SPROW 1032
#define KVROW 72     // k-chunk row stride (144B, 16B-aligned)
#define VTROW 136    // v^T row stride (272B, 16B-aligned)
#define KVBUF (128 * KVROW)
#define NEGBIG (-1.0e30f)

#define ATTN_SMEM_SS   (16 * SSROW * 4)
#define ATTN_SMEM_SPB  (16 * SPROW * 2)
#define ATTN_SMEM_SKV  (2 * KVBUF * 2)
#define ATTN_SMEM_RED  (4 * 256 * 4)
#define ATTN_SMEM_SIZE (ATTN_SMEM_SS + ATTN_SMEM_SPB + ATTN_SMEM_SKV + ATTN_SMEM_RED)

__global__ __launch_bounds__(256) void attn_kernel(
    const uint16_t* __restrict__ Qb, const uint16_t* __restrict__ Kb,
    const uint16_t* __restrict__ Vb, const uint8_t* __restrict__ Mask,
    const float* __restrict__ Qmask, float* __restrict__ Attn,
    uint16_t* __restrict__ Xws)
{
  extern __shared__ char smem_raw[];
  float*    sS   = (float*)smem_raw;
  uint16_t* sPb  = (uint16_t*)(smem_raw + ATTN_SMEM_SS);
  uint16_t* sKV  = (uint16_t*)(smem_raw + ATTN_SMEM_SS + ATTN_SMEM_SPB);
  float*    sred = (float*)(smem_raw + ATTN_SMEM_SS + ATTN_SMEM_SPB + ATTN_SMEM_SKV);

  const int tid = threadIdx.x;
  const int w = tid >> 5, lane = tid & 31;
  const int half = lane >> 4, l15 = lane & 15;
  const int hb = blockIdx.y;            // h*4 + b
  const int h = hb >> 2, bi = hb & 3;
  const int i0 = blockIdx.x * 16;

  // ---- Phase 1: S = q @ k^T (scaled), fp32 into sS ----
  FragAB a0, a1;
  {
    const uint4* q4 = (const uint4*)(
        Qb + ((size_t)(bi * 1024 + i0 + l15) * 1024 + h * 64));
    a0.q[0] = q4[half];
    a0.q[1] = q4[2 + half];
    a1.q[0] = q4[4 + half];
    a1.q[1] = q4[6 + half];
  }
  auto stageK = [&](int cc, int buf) {
    const int j0 = cc * 128;
    uint16_t* dst = sKV + buf * KVBUF;
#pragma unroll
    for (int it = 0; it < 4; ++it) {
      int idx = tid + it * 256;
      int row = idx >> 3, q = idx & 7;
      const uint4* src = (const uint4*)(
          Kb + ((size_t)(bi * 1024 + j0 + row)) * 1024 + h * 64);
      *(uint4*)((uint32_t*)&dst[row * KVROW] + q * 4) = src[q];
    }
  };
  auto qk_chunk = [&](int c, int buf) {
    const uint16_t* krow = &sKV[buf * KVBUF + (w * 16 + l15) * KVROW];
    FragAB b0, b1;
    const uint4* r = (const uint4*)krow;
    b0.q[0] = r[half];
    b0.q[1] = r[2 + half];
    b1.q[0] = r[4 + half];
    b1.q[1] = r[6 + half];
    v8f acc = {};
    acc = wmma_bf16(a0, b0, acc);
    acc = wmma_bf16(a1, b1, acc);
    const int jg = c * 128 + w * 16 + l15;
#pragma unroll
    for (int r8 = 0; r8 < 8; ++r8)
      sS[(r8 + 8 * half) * SSROW + jg] = acc[r8] * 0.125f;
  };

  stageK(0, 0);
  __syncthreads();
  for (int c = 0; c < 7; ++c) {
    stageK(c + 1, (c & 1) ^ 1);   // overlap staging with WMMA of chunk c
    qk_chunk(c, c & 1);
    __syncthreads();
  }
  qk_chunk(7, 1);
  __syncthreads();

  // ---- Phase 2: masked softmax per row (fp32); write attn; probs -> bf16 ----
  {
    const int il = 2 * w + half;
    const int ig = i0 + il;
    const uint8_t* mrow = Mask + ((size_t)hb * 1024 + ig) * 1024;
    float* srow = sS + (size_t)il * SSROW;
    float mx = NEGBIG;
    for (int c = l15; c < 1024; c += 16) {
      float s = srow[c];
      if (mrow[c]) { s = NEGBIG; srow[c] = s; }
      mx = fmaxf(mx, s);
    }
#pragma unroll
    for (int off = 8; off > 0; off >>= 1) mx = fmaxf(mx, __shfl_xor(mx, off, 16));
    float sum = 0.0f;
    for (int c = l15; c < 1024; c += 16) {
      float e = __expf(srow[c] - mx);
      sum += e;
      srow[c] = mrow[c] ? 0.0f : e;       // masked contribute to sum, output 0
    }
#pragma unroll
    for (int off = 8; off > 0; off >>= 1) sum += __shfl_xor(sum, off, 16);
    const float inv = Qmask[bi * 1024 + ig] / sum;
    float* arow = Attn + ((size_t)hb * 1024 + ig) * 1024;
    for (int c = l15; c < 1024; c += 16) {
      float p = srow[c] * inv;
      arow[c] = p;                        // fp32-exact attn output
      sPb[il * SPROW + c] = f32_to_bf16(p);
    }
  }
  __syncthreads();

  // ---- Phase 3: x = P(16x1024) @ v(1024x64), K split across wave pairs ----
  const int ntile = w >> 1;
  const int khalf = w & 1;

  auto stageV = [&](int cc, int buf) {
    const int j0 = cc * 128;
    uint16_t* dst = sKV + buf * KVBUF;
#pragma unroll
    for (int it = 0; it < 4; ++it) {
      int idx = tid + it * 256;
      int j = idx >> 3, q = idx & 7;      // q-th uint4 covers d = 8q..8q+7
      uint4 v4 = *(const uint4*)(
          Vb + ((size_t)(bi * 1024 + j0 + j)) * 1024 + h * 64 + q * 8);
      uint32_t vv[4] = {v4.x, v4.y, v4.z, v4.w};
#pragma unroll
      for (int e = 0; e < 4; ++e) {
        int d = q * 8 + 2 * e;
        dst[(d + 0) * VTROW + j] = (uint16_t)(vv[e] & 0xFFFFu);
        dst[(d + 1) * VTROW + j] = (uint16_t)(vv[e] >> 16);
      }
    }
  };
  v8f xacc = {};
  auto pv_chunk = [&](int c, int buf) {   // caller guarantees wave-uniform call
    const uint4* prow = (const uint4*)&sPb[l15 * SPROW];
    const uint4* vrow =
        (const uint4*)&sKV[buf * KVBUF + (ntile * 16 + l15) * VTROW];
#pragma unroll
    for (int sub = 0; sub < 4; ++sub) {
      FragAB a, b;
      const int kq = (c * 128 + sub * 32) >> 3;
      a.q[0] = prow[kq + half];
      a.q[1] = prow[kq + 2 + half];
      b.q[0] = vrow[sub * 4 + half];
      b.q[1] = vrow[sub * 4 + 2 + half];
      xacc = wmma_bf16(a, b, xacc);
    }
  };

  stageV(0, 0);
  __syncthreads();
  for (int c = 0; c < 7; ++c) {
    if ((c >> 2) == khalf) pv_chunk(c, c & 1);
    stageV(c + 1, (c & 1) ^ 1);
    __syncthreads();
  }
  if (khalf == 1) pv_chunk(7, 1);

  if (khalf == 1) {
#pragma unroll
    for (int r = 0; r < 8; ++r) sred[(w >> 1) * 256 + lane * 8 + r] = xacc[r];
  }
  __syncthreads();
  if (khalf == 0) {
    const int d = ntile * 16 + l15;
#pragma unroll
    for (int r = 0; r < 8; ++r) {
      float val = xacc[r] + sred[(w >> 1) * 256 + lane * 8 + r];
      int il = r + 8 * half;
      Xws[((size_t)(bi * 1024 + i0 + il)) * 1024 + h * 64 + d] =
          f32_to_bf16(val);
    }
  }
}

// ---------------------------------------------------------------------------
// Kernel 3: out = [x | query] @ Wf + bf + query   (K = 2048, chunk-aligned
// source switch at k=1024), double-buffered, last iteration peeled
// ---------------------------------------------------------------------------
#define FNCH 64   // 2048 / KC

__global__ __launch_bounds__(256) void final_gemm(
    const uint16_t* __restrict__ Xws, const float* __restrict__ Q,
    const float* __restrict__ Wf, const float* __restrict__ bf_,
    float* __restrict__ Out)
{
  __shared__ uint16_t ldsA[2][64][KPAD];
  __shared__ uint16_t ldsBt[2][64][KPAD];
  const int tid = threadIdx.x;
  const int w = tid >> 5, lane = tid & 31;
  const int half = lane >> 4, l15 = lane & 15;
  const int bm = blockIdx.x * 64, bn = blockIdx.y * 64;
  const int mt = w & 3, np = w >> 2;

  const int sm = tid >> 2, skq = (tid & 3) * 8;

  auto stage = [&](int k0, int buf) {
    {
      uint4 p;
      if (k0 < 1024) {
        p = *(const uint4*)&Xws[(size_t)(bm + sm) * 1024 + k0 + skq];
      } else {
        const float4* src =
            (const float4*)&Q[(size_t)(bm + sm) * 1024 + (k0 - 1024) + skq];
        float4 x0 = src[0], x1 = src[1];
        p = uint4{ pack_bf16(x0.x, x0.y), pack_bf16(x0.z, x0.w),
                   pack_bf16(x1.x, x1.y), pack_bf16(x1.z, x1.w) };
      }
      *(uint4*)&ldsA[buf][sm][skq] = p;
    }
#pragma unroll
    for (int it = 0; it < 2; ++it) {
      int idx = tid + it * 256;
      int k = idx >> 4, nq = (idx & 15) * 4;
      float4 wv = *(const float4*)&Wf[(size_t)(k0 + k) * 1024 + bn + nq];
      uint32_t p01 = pack_bf16(wv.x, wv.y), p23 = pack_bf16(wv.z, wv.w);
      ldsBt[buf][nq + 0][k] = (uint16_t)p01;
      ldsBt[buf][nq + 1][k] = (uint16_t)(p01 >> 16);
      ldsBt[buf][nq + 2][k] = (uint16_t)p23;
      ldsBt[buf][nq + 3][k] = (uint16_t)(p23 >> 16);
    }
  };

  v8f acc0 = {}; v8f acc1 = {};
  stage(0, 0);
  __syncthreads();
  for (int c = 0; c < FNCH - 1; ++c) {
    const int buf = c & 1;
    FragAB a, b0, b1;
    load_fragA(a, &ldsA[buf][mt * 16 + l15][0], half);
    load_fragB(b0, &ldsBt[buf][(np * 2 + 0) * 16 + l15][0], half);
    load_fragB(b1, &ldsBt[buf][(np * 2 + 1) * 16 + l15][0], half);
    stage((c + 1) * KC, buf ^ 1);
    acc0 = wmma_bf16(a, b0, acc0);
    acc1 = wmma_bf16(a, b1, acc1);
    __syncthreads();
  }
  {
    const int buf = (FNCH - 1) & 1;
    FragAB a, b0, b1;
    load_fragA(a, &ldsA[buf][mt * 16 + l15][0], half);
    load_fragB(b0, &ldsBt[buf][(np * 2 + 0) * 16 + l15][0], half);
    load_fragB(b1, &ldsBt[buf][(np * 2 + 1) * 16 + l15][0], half);
    acc0 = wmma_bf16(a, b0, acc0);
    acc1 = wmma_bf16(a, b1, acc1);
  }
#pragma unroll
  for (int t = 0; t < 2; ++t) {
    v8f acc = (t == 0) ? acc0 : acc1;
    int n = bn + (np * 2 + t) * 16 + l15;
    float bv = bf_[n];
#pragma unroll
    for (int r = 0; r < 8; ++r) {
      int m = bm + mt * 16 + r + 8 * half;
      Out[(size_t)m * 1024 + n] = acc[r] + bv + Q[(size_t)m * 1024 + n];
    }
  }
}

// ---------------------------------------------------------------------------
extern "C" void kernel_launch(void* const* d_in, const int* in_sizes, int n_in,
                              void* d_out, int out_size, void* d_ws,
                              size_t ws_size, hipStream_t stream) {
  const float*   query = (const float*)d_in[0];
  const float*   key_  = (const float*)d_in[1];
  const float*   value = (const float*)d_in[2];
  const uint8_t* mask  = (const uint8_t*)d_in[3];
  const float*   qmask = (const float*)d_in[4];
  const float*   Wq = (const float*)d_in[5];
  const float*   bq = (const float*)d_in[6];
  const float*   Wk = (const float*)d_in[7];
  const float*   bk = (const float*)d_in[8];
  const float*   Wv = (const float*)d_in[9];
  const float*   bv = (const float*)d_in[10];
  const float*   Wf = (const float*)d_in[11];
  const float*   bf = (const float*)d_in[12];

  const size_t MN = (size_t)4096 * 1024;
  uint16_t* qb = (uint16_t*)d_ws;
  uint16_t* kb = qb + MN;
  uint16_t* vb = kb + MN;
  uint16_t* xw = vb + MN;

  float* out  = (float*)d_out;
  float* attn = out + MN;   // out (4,1024,1024) then attn (64,1024,1024)

  (void)hipFuncSetAttribute((const void*)attn_kernel,
                            hipFuncAttributeMaxDynamicSharedMemorySize,
                            ATTN_SMEM_SIZE);

  dim3 blk(256);
  proj_gemm<<<dim3(64, 16), blk, 0, stream>>>(query, Wq, bq, qb);
  proj_gemm<<<dim3(64, 16), blk, 0, stream>>>(key_,  Wk, bk, kb);
  proj_gemm<<<dim3(64, 16), blk, 0, stream>>>(value, Wv, bv, vb);
  attn_kernel<<<dim3(64, 64), blk, ATTN_SMEM_SIZE, stream>>>(qb, kb, vb, mask, qmask, attn, xw);
  final_gemm<<<dim3(64, 16), blk, 0, stream>>>(xw, query, Wf, bf, out);
}
